// Attention_67662914781482
// MI455X (gfx1250) — compile-verified, matrix operand order checked by reading, not run on hardware
//
#include <hip/hip_runtime.h>

#define BB 2
#define SS 2048
#define DD 2048
#define HH 16
#define HD 128

typedef __attribute__((ext_vector_type(16))) _Float16 v16h;
typedef __attribute__((ext_vector_type(8)))  float    v8f;

union FragU { v16h v; uint4 q[2]; };

// Load a 16x32 f16 WMMA A/B fragment (row-major, K-contiguous source).
// Per ISA 7.12.2: lanes 0-15 hold K {0..7,16..23}, lanes 16-31 hold K {8..15,24..31}.
// Caller passes p = base + row*ldk + half*8 ; we fetch p[0..7] and p[16..23].
__device__ __forceinline__ v16h load_frag(const _Float16* p) {
  FragU u;
  u.q[0] = *reinterpret_cast<const uint4*>(p);
  u.q[1] = *reinterpret_cast<const uint4*>(p + 16);
  return u.v;
}

__device__ __forceinline__ v8f zero8() {
  v8f z = {0.f,0.f,0.f,0.f,0.f,0.f,0.f,0.f};
  return z;
}

// ---------------- fp32 -> fp16 conversion ----------------
__global__ __launch_bounds__(256) void cvt_f32_f16(const float* __restrict__ in,
                                                   _Float16* __restrict__ out, int n) {
  int i = blockIdx.x * blockDim.x + threadIdx.x;
  if (i < n) out[i] = (_Float16)in[i];
}

// ---------------- RoPE + fp16 conversion (scale folds 1/sqrt(HD) for Q) ----------------
__global__ __launch_bounds__(256) void rope_cvt(const float* __restrict__ qf,
                                                const float* __restrict__ cosb,
                                                const float* __restrict__ sinb,
                                                _Float16* __restrict__ out, float scale) {
  size_t i = (size_t)blockIdx.x * blockDim.x + threadIdx.x;
  const size_t total = (size_t)BB * SS * DD;
  if (i >= total) return;
  int col = (int)(i % DD);
  size_t bs = i / DD;
  int hd = col & (HD - 1);
  float c  = cosb[bs * HD + hd];
  float sn = sinb[bs * HD + hd];
  float v  = qf[i];
  float rot = (hd < HD / 2) ? -qf[i + HD / 2] : qf[i - HD / 2];
  out[i] = (_Float16)((v * c + rot * sn) * scale);
}

// ---------------- WMMA GEMM: C[M,N] = A[M,K] * W[N,K]^T ----------------
// mode 0: write f32 row-major to Cf.
// mode 1: write f16 to CvT in (B,H,HD,S) transposed layout (for V).
__global__ __launch_bounds__(256) void gemm_xwT(const _Float16* __restrict__ A,
                                                const _Float16* __restrict__ W,
                                                float* __restrict__ Cf,
                                                _Float16* __restrict__ CvT,
                                                int M, int N, int K, int mode) {
  const int lane = threadIdx.x & 31;
  const int wid  = threadIdx.x >> 5;   // 0..7
  const int wm   = wid & 3;            // 4 M-slices of 32 rows
  const int wn   = wid >> 2;           // 2 N-slices of 64 cols
  const int m0   = blockIdx.y * 128 + wm * 32;
  const int n0   = blockIdx.x * 128 + wn * 64;
  const int r    = lane & 15;
  const int hf   = lane >> 4;

  v8f acc[2][4];
  for (int i = 0; i < 2; i++)
    for (int j = 0; j < 4; j++) acc[i][j] = zero8();

  const _Float16* arow0 = A + (size_t)(m0 + r) * K + hf * 8;
  const _Float16* arow1 = A + (size_t)(m0 + 16 + r) * K + hf * 8;
  const _Float16* brow[4];
  for (int j = 0; j < 4; j++) brow[j] = W + (size_t)(n0 + j * 16 + r) * K + hf * 8;

  for (int k0 = 0; k0 < K; k0 += 32) {
    v16h a0 = load_frag(arow0 + k0);
    v16h a1 = load_frag(arow1 + k0);
    v16h b[4];
    for (int j = 0; j < 4; j++) b[j] = load_frag(brow[j] + k0);
    for (int j = 0; j < 4; j++) {
      acc[0][j] = __builtin_amdgcn_wmma_f32_16x16x32_f16(false, a0, false, b[j], (short)0, acc[0][j], false, false);
      acc[1][j] = __builtin_amdgcn_wmma_f32_16x16x32_f16(false, a1, false, b[j], (short)0, acc[1][j], false, false);
    }
  }

  // C/D layout: lane holds column n = lane&15 ; rows m = 8*hf + r for VGPR r.
  if (mode == 0) {
    for (int i = 0; i < 2; i++)
      for (int j = 0; j < 4; j++) {
        int nn = n0 + j * 16 + r;
        int mb = m0 + i * 16 + hf * 8;
        for (int rr = 0; rr < 8; rr++)
          Cf[(size_t)(mb + rr) * N + nn] = acc[i][j][rr];
      }
  } else {
    for (int i = 0; i < 2; i++)
      for (int j = 0; j < 4; j++) {
        int nn = n0 + j * 16 + r;
        int h  = nn / HD, hd = nn % HD;
        int mb = m0 + i * 16 + hf * 8;
        for (int rr = 0; rr < 8; rr++) {
          int m = mb + rr;
          int b_ = m / SS, s = m % SS;
          CvT[((size_t)(b_ * HH + h) * HD + hd) * SS + s] = (_Float16)acc[i][j][rr];
        }
      }
  }
}

// ---------------- Flash attention ----------------
// Q,Kr: (B,S,D) f16 (RoPE applied; Q pre-scaled by 1/sqrt(HD))
// Vt:   (B,H,HD,S) f16 transposed
// O:    (B,S,D) f16
// Block = 128 threads = 4 waves; each wave owns 16 query rows.
__global__ __launch_bounds__(128) void flash_attn(const _Float16* __restrict__ Q,
                                                  const _Float16* __restrict__ Kr,
                                                  const _Float16* __restrict__ Vt,
                                                  const float* __restrict__ mask,
                                                  _Float16* __restrict__ O) {
  __shared__ _Float16 plds[4][16][32];   // per-wave P staging tile
  const int lane = threadIdx.x & 31;
  const int wid  = threadIdx.x >> 5;
  const int r    = lane & 15;
  const int hf   = lane >> 4;
  const int bh   = blockIdx.y;
  const int b    = bh / HH, h = bh % HH;
  const int q0   = blockIdx.x * 64 + wid * 16;

  // Q fragments: 16 rows x 128 K -> 4 fragments, pinned in registers.
  v16h qf[4];
  const _Float16* qbase = Q + (size_t)(b * SS + q0 + r) * DD + h * HD + hf * 8;
  for (int kc = 0; kc < 4; kc++) qf[kc] = load_frag(qbase + kc * 32);

  float mrow[8], lrow[8];
  v8f o[8];
  for (int rr = 0; rr < 8; rr++) { mrow[rr] = -1e30f; lrow[rr] = 0.f; }
  for (int n = 0; n < 8; n++) o[n] = zero8();

  const _Float16* kbase = Kr + (size_t)(b * SS) * DD + h * HD;
  const _Float16* vbase = Vt + (size_t)(b * HH + h) * HD * SS;

  for (int kb = 0; kb < SS; kb += 32) {
    // ---- scores for 32 key positions (two 16x16 f32 tiles) ----
    v8f sfr[2];
    for (int nt = 0; nt < 2; nt++) {
      v8f s = zero8();
      const _Float16* kp = kbase + (size_t)(kb + nt * 16 + r) * DD + hf * 8;
      for (int kc = 0; kc < 4; kc++) {
        v16h kf = load_frag(kp + kc * 32);
        s = __builtin_amdgcn_wmma_f32_16x16x32_f16(false, qf[kc], false, kf, (short)0, s, false, false);
      }
      float mk = mask[b * SS + kb + nt * 16 + r];   // column-indexed mask
      for (int rr = 0; rr < 8; rr++) s[rr] += mk;
      sfr[nt] = s;
    }

    // ---- online softmax (rows live as 8 regs/lane, uniform across 16-lane halves) ----
    float rm[8];
    for (int rr = 0; rr < 8; rr++) rm[rr] = fmaxf(sfr[0][rr], sfr[1][rr]);
    for (int msk = 1; msk < 16; msk <<= 1)
      for (int rr = 0; rr < 8; rr++)
        rm[rr] = fmaxf(rm[rr], __shfl_xor(rm[rr], msk, 32));

    float scl[8], rsum[8];
    for (int rr = 0; rr < 8; rr++) {
      float mnew = fmaxf(mrow[rr], rm[rr]);
      scl[rr]  = __expf(mrow[rr] - mnew);
      mrow[rr] = mnew;
      float e0 = __expf(sfr[0][rr] - mnew);
      float e1 = __expf(sfr[1][rr] - mnew);
      sfr[0][rr] = e0; sfr[1][rr] = e1;
      rsum[rr] = e0 + e1;
    }
    for (int msk = 1; msk < 16; msk <<= 1)
      for (int rr = 0; rr < 8; rr++)
        rsum[rr] += __shfl_xor(rsum[rr], msk, 32);
    for (int rr = 0; rr < 8; rr++) lrow[rr] = lrow[rr] * scl[rr] + rsum[rr];
    for (int n = 0; n < 8; n++)
      for (int rr = 0; rr < 8; rr++) o[n][rr] *= scl[rr];

    // ---- stage P (C-layout) into LDS, reload as A-fragment (same-wave DS is in-order) ----
    for (int nt = 0; nt < 2; nt++)
      for (int rr = 0; rr < 8; rr++)
        plds[wid][hf * 8 + rr][nt * 16 + r] = (_Float16)sfr[nt][rr];
    v16h pf = load_frag(&plds[wid][r][hf * 8]);

    // ---- O += P (16x32) * V (32x128) ----
    for (int n = 0; n < 8; n++) {
      const _Float16* vp = vbase + (size_t)(n * 16 + r) * SS + kb + hf * 8;
      v16h vf = load_frag(vp);
      o[n] = __builtin_amdgcn_wmma_f32_16x16x32_f16(false, pf, false, vf, (short)0, o[n], false, false);
    }
  }

  // ---- normalize and store (B,S,D) f16 ----
  _Float16* obase = O + (size_t)(b * SS + q0) * DD + h * HD;
  for (int rr = 0; rr < 8; rr++) {
    float inv = 1.0f / lrow[rr];
    int m = hf * 8 + rr;
    for (int n = 0; n < 8; n++)
      obase[(size_t)m * DD + n * 16 + r] = (_Float16)(o[n][rr] * inv);
  }
}

extern "C" void kernel_launch(void* const* d_in, const int* in_sizes, int n_in,
                              void* d_out, int out_size, void* d_ws, size_t ws_size,
                              hipStream_t stream) {
  (void)in_sizes; (void)n_in; (void)out_size; (void)ws_size;
  const float* x    = (const float*)d_in[0];
  const float* cosb = (const float*)d_in[1];
  const float* sinb = (const float*)d_in[2];
  const float* mask = (const float*)d_in[3];
  const float* wq   = (const float*)d_in[4];
  const float* wk   = (const float*)d_in[5];
  const float* wv   = (const float*)d_in[6];
  const float* wo   = (const float*)d_in[7];
  float* out = (float*)d_out;

  const size_t nX = (size_t)BB * SS * DD;   // 8,388,608
  const size_t nW = (size_t)DD * DD;        // 4,194,304

  char* p = (char*)d_ws;
  _Float16* x16  = (_Float16*)p; p += nX * 2;
  _Float16* wq16 = (_Float16*)p; p += nW * 2;
  _Float16* wk16 = (_Float16*)p; p += nW * 2;
  _Float16* wv16 = (_Float16*)p; p += nW * 2;
  _Float16* wo16 = (_Float16*)p; p += nW * 2;
  _Float16* q16  = (_Float16*)p; p += nX * 2;
  _Float16* k16  = (_Float16*)p; p += nX * 2;
  _Float16* vT16 = (_Float16*)p; p += nX * 2;
  float*    f32s = (float*)p;               // nX * 4 bytes f32 scratch
  _Float16* o16  = (_Float16*)p;            // aliases f32s (dead after RoPE(K))

  dim3 blk256(256), blk128(128);
  dim3 gCvtX((unsigned)((nX + 255) / 256));
  dim3 gCvtW((unsigned)((nW + 255) / 256));
  dim3 gGemm(DD / 128, (BB * SS) / 128);    // (16, 32)
  dim3 gFlash(SS / 64, BB * HH);            // (32, 32)

  // fp16 copies
  cvt_f32_f16<<<gCvtX, blk256, 0, stream>>>(x,  x16,  (int)nX);
  cvt_f32_f16<<<gCvtW, blk256, 0, stream>>>(wq, wq16, (int)nW);
  cvt_f32_f16<<<gCvtW, blk256, 0, stream>>>(wk, wk16, (int)nW);
  cvt_f32_f16<<<gCvtW, blk256, 0, stream>>>(wv, wv16, (int)nW);
  cvt_f32_f16<<<gCvtW, blk256, 0, stream>>>(wo, wo16, (int)nW);

  // Q projection + RoPE (folds 1/sqrt(HD))
  gemm_xwT<<<gGemm, blk256, 0, stream>>>(x16, wq16, f32s, nullptr, BB * SS, DD, DD, 0);
  rope_cvt<<<gCvtX, blk256, 0, stream>>>(f32s, cosb, sinb, q16, 0.08838834764831845f);

  // K projection + RoPE
  gemm_xwT<<<gGemm, blk256, 0, stream>>>(x16, wk16, f32s, nullptr, BB * SS, DD, DD, 0);
  rope_cvt<<<gCvtX, blk256, 0, stream>>>(f32s, cosb, sinb, k16, 1.0f);

  // V projection, written directly transposed (B,H,HD,S) f16
  gemm_xwT<<<gGemm, blk256, 0, stream>>>(x16, wv16, nullptr, vT16, BB * SS, DD, DD, 1);

  // Flash attention -> o16 (B,S,D) f16
  flash_attn<<<gFlash, blk128, 0, stream>>>(q16, k16, vT16, mask, o16);

  // Output projection -> d_out fp32
  gemm_xwT<<<gGemm, blk256, 0, stream>>>(o16, wo16, out, nullptr, BB * SS, DD, DD, 0);
}
